// MyLSTM_25666724561019
// MI455X (gfx1250) — matrix operand order repaired and checked by reading, not compiled, not run on hardware
//
#include <hip/hip_runtime.h>

#define T_STEPS 512
#define BATCH   64
#define IN_DIM  512
#define HID     1024
#define GATES   4096   // 4*HID

#define KCHUNK   128                        // K elements staged per chunk
#define NCHUNK_X (IN_DIM / KCHUNK)          // 4
#define NCHUNK_H (HID / KCHUNK)             // 8
#define NCHUNK   (NCHUNK_X + NCHUNK_H)      // 12
#define A_STRIDE 136                        // 128 + 8 bf16 pad (4-bank rotate/row)

typedef __bf16 bf16_t;
typedef __attribute__((ext_vector_type(16))) __bf16 v16bf;
typedef __attribute__((ext_vector_type(8)))  __bf16 v8bf;
typedef __attribute__((ext_vector_type(8)))  float  v8f;

// ---------------------------------------------------------------------------
// A-fragment (16x32 MxK bf16) from LDS-staged tile (row stride A_STRIDE).
// lane<16: row m0+lane, K = kk+0..15; lane>=16: row m0+lane-16,
// K = kk+16..31.
// ---------------------------------------------------------------------------
__device__ __forceinline__ v16bf load_frag_a_lds(const bf16_t* atile,
                                                 int m0, int kk, int lane) {
    const int hf = lane >> 4;
    const int mr = lane & 15;
    const bf16_t* p = atile + (m0 + mr) * A_STRIDE + kk + 16 * hf;
    v8bf lo = *(const v8bf*)(p);
    v8bf hi = *(const v8bf*)(p + 8);
    v16bf r;
#pragma unroll
    for (int i = 0; i < 8; ++i) { r[i] = lo[i]; r[i + 8] = hi[i]; }
    return r;
}

// ---------------------------------------------------------------------------
// B-fragment (32x16 KxN) from fragment-swizzled weights: fragment
// f = (g0/16)*ktiles + k0/32 stored as 32 lane-records of 16 bf16, so the
// wave reads one contiguous 1KB burst.
// ---------------------------------------------------------------------------
__device__ __forceinline__ v16bf load_frag_b_sw(const bf16_t* __restrict__ wsw,
                                                int ktiles, int g0, int k0,
                                                int lane) {
    const bf16_t* p = wsw +
        ((((size_t)(g0 >> 4) * ktiles + (k0 >> 5)) << 5) + lane) * 16;
    v8bf lo = *(const v8bf*)(p);
    v8bf hi = *(const v8bf*)(p + 8);
    v16bf r;
#pragma unroll
    for (int i = 0; i < 8; ++i) { r[i] = lo[i]; r[i + 8] = hi[i]; }
    return r;
}

// ---------------------------------------------------------------------------
// Prep: x fp32->bf16; weights fp32->bf16 *fragment-swizzled*; fused bias;
// zero initial state.
// swizzled[((nt*ktiles+kt)*32+lane)*16+e] = W[nt*16+(lane&15)][kt*32+(lane>>4)*16+e]
// ---------------------------------------------------------------------------
__global__ void lstm_prep(const float* __restrict__ x,
                          const float* __restrict__ wih_f,
                          const float* __restrict__ bih,
                          const float* __restrict__ whh_f,
                          const float* __restrict__ bhh,
                          bf16_t* __restrict__ xb,
                          bf16_t* __restrict__ wib,
                          bf16_t* __restrict__ whb,
                          float*  __restrict__ bias,
                          float*  __restrict__ c,
                          bf16_t* __restrict__ h0b) {
    const size_t tid = (size_t)blockIdx.x * blockDim.x + threadIdx.x;
    const size_t np  = (size_t)gridDim.x * blockDim.x;

    for (size_t i = tid; i < (size_t)T_STEPS * BATCH * IN_DIM; i += np)
        xb[i] = (bf16_t)x[i];

    {   // W_ih swizzle, ktiles = IN_DIM/32
        const size_t KT = IN_DIM / 32;
        for (size_t i = tid; i < (size_t)GATES * IN_DIM; i += np) {
            size_t e    = i & 15;
            size_t lane = (i >> 4) & 31;
            size_t f    = i >> 9;
            size_t kt   = f % KT;
            size_t nt   = f / KT;
            size_t row  = nt * 16 + (lane & 15);
            size_t col  = kt * 32 + (lane >> 4) * 16 + e;
            wib[i] = (bf16_t)wih_f[row * IN_DIM + col];
        }
    }
    {   // W_hh swizzle, ktiles = HID/32
        const size_t KT = HID / 32;
        for (size_t i = tid; i < (size_t)GATES * HID; i += np) {
            size_t e    = i & 15;
            size_t lane = (i >> 4) & 31;
            size_t f    = i >> 9;
            size_t kt   = f % KT;
            size_t nt   = f / KT;
            size_t row  = nt * 16 + (lane & 15);
            size_t col  = kt * 32 + (lane >> 4) * 16 + e;
            whb[i] = (bf16_t)whh_f[row * HID + col];
        }
    }
    for (size_t i = tid; i < (size_t)GATES; i += np)
        bias[i] = bih[i] + bhh[i];
    for (size_t i = tid; i < (size_t)BATCH * HID; i += np) {
        c[i]   = 0.0f;
        h0b[i] = (bf16_t)0.0f;
    }
}

// ---------------------------------------------------------------------------
// One LSTM timestep. Grid = HID/16 = 64 blocks, 256 threads = 8 waves.
// Wave w: gate q = w&3 (i,f,g,o), row-half mh = w>>2. A operand (x[t] then h)
// is async-staged into double-buffered LDS (12 chunks of 64xKCHUNK bf16);
// B fragments stream coalesced from L2-resident swizzled weights.
// ---------------------------------------------------------------------------
__global__ __launch_bounds__(256)
void lstm_step(const bf16_t* __restrict__ xb,     // [T,B,I] bf16
               const bf16_t* __restrict__ wih,    // swizzled [4096x512]
               const bf16_t* __restrict__ whh,    // swizzled [4096x1024]
               const float*  __restrict__ bias,   // [4096] = b_ih + b_hh
               const bf16_t* __restrict__ hprev,  // [B,H] bf16
               bf16_t* __restrict__ hnext,        // [B,H] bf16
               float*  __restrict__ c,            // [B,H] fp32 (in/out)
               float*  __restrict__ out,          // [T,B,H] fp32
               int t) {
    __shared__ bf16_t atile[2][BATCH][A_STRIDE];  // ~34 KB, double-buffered
    __shared__ float  gbuf[4][BATCH][17];         // ~17 KB, gate exchange

    const int tid  = threadIdx.x;
    const int lane = tid & 31;
    const int wave = tid >> 5;          // 0..7
    const int q    = wave & 3;          // gate: 0=i 1=f 2=g 3=o
    const int mh   = wave >> 2;         // 0/1
    const int n0   = blockIdx.x * 16;   // h-column tile
    const int grow = q * HID + n0;      // gate row base in [0,4096)
    const int mbase = mh * 32;

    const bf16_t* xt = xb + (size_t)t * BATCH * IN_DIM;
    const uint32_t abase = (uint32_t)(uintptr_t)(&atile[0][0][0]);

    // ---- async stage of one 64 x KCHUNK A-chunk into LDS buffer (p&1) ----
    auto stage = [&](int p) {
        const bf16_t* src;
        int ld, kc;
        if (p < NCHUNK_X) { src = xt;    ld = IN_DIM; kc = p * KCHUNK; }
        else              { src = hprev; ld = HID;    kc = (p - NCHUNK_X) * KCHUNK; }
        const uint32_t bbase = abase + (uint32_t)((p & 1) * BATCH * A_STRIDE * 2);
#pragma unroll
        for (int j = 0; j < 4; ++j) {                  // 4 asyncs/wave/chunk
            const int u   = tid + 256 * j;             // 0..1023
            const int row = u >> 4;
            const int c16 = u & 15;
            const bf16_t* g = src + (size_t)row * ld + kc + c16 * 8;
            const uint32_t loff =
                bbase + (uint32_t)(row * A_STRIDE + c16 * 8) * 2;
            asm volatile("global_load_async_to_lds_b128 %0, %1, off"
                         :: "v"(loff), "v"(g) : "memory");
        }
    };

    v8f acc0 = {};   // rows mbase + 0..15
    v8f acc1 = {};   // rows mbase + 16..31

    stage(0);
    for (int p = 0; p < NCHUNK; ++p) {
        if (p + 1 < NCHUNK) {
            stage(p + 1);
            asm volatile("s_wait_asynccnt 0x4" ::: "memory");  // chunk p landed
        } else {
            asm volatile("s_wait_asynccnt 0x0" ::: "memory");
        }
        __syncthreads();                                        // all waves' p in LDS

        const bf16_t* abuf   = &atile[p & 1][0][0];
        const bf16_t* wsw    = (p < NCHUNK_X) ? wih : whh;
        const int     ktiles = (p < NCHUNK_X) ? (IN_DIM / 32) : (HID / 32);
        const int     kcw    = (p < NCHUNK_X) ? p * KCHUNK
                                              : (p - NCHUNK_X) * KCHUNK;
#pragma unroll
        for (int kk = 0; kk < KCHUNK; kk += 32) {
            v16bf b  = load_frag_b_sw(wsw, ktiles, grow, kcw + kk, lane);
            v16bf a0 = load_frag_a_lds(abuf, mbase,      kk, lane);
            v16bf a1 = load_frag_a_lds(abuf, mbase + 16, kk, lane);
            acc0 = __builtin_amdgcn_wmma_f32_16x16x32_bf16(false, a0, false, b,
                                                           (short)0, acc0, false, false);
            acc1 = __builtin_amdgcn_wmma_f32_16x16x32_bf16(false, a1, false, b,
                                                           (short)0, acc1, false, false);
        }
        __syncthreads();                 // done reading before buffer reuse
    }

    // ---- bias + activation, spill to LDS (C/D: M = r + 8*(lane>=16), N=lane%16)
    const int hf   = lane >> 4;
    const int ncol = lane & 15;
    const float bv = bias[grow + ncol];
#pragma unroll
    for (int r = 0; r < 8; ++r) {
        float v0 = acc0[r] + bv;
        float v1 = acc1[r] + bv;
        if (q == 2) {                      // cell gate: tanh
            v0 = tanhf(v0);
            v1 = tanhf(v1);
        } else {                           // i/f/o: sigmoid
            v0 = 1.0f / (1.0f + __expf(-v0));
            v1 = 1.0f / (1.0f + __expf(-v1));
        }
        gbuf[q][mbase + 8 * hf + r][ncol]      = v0;
        gbuf[q][mbase + 16 + 8 * hf + r][ncol] = v1;
    }
    __syncthreads();

    // ---- fused cell update: c' = f*c + i*g ; h = o*tanh(c') ----
#pragma unroll
    for (int j = 0; j < 4; ++j) {
        const int el  = tid + 256 * j;     // 64x16 tile
        const int row = el >> 4;
        const int col = el & 15;
        const float iv = gbuf[0][row][col];
        const float fv = gbuf[1][row][col];
        const float gv = gbuf[2][row][col];
        const float ov = gbuf[3][row][col];
        const size_t hidx = (size_t)row * HID + n0 + col;
        const float cn = fv * c[hidx] + iv * gv;
        const float hv = ov * tanhf(cn);
        c[hidx] = cn;
        out[(size_t)t * BATCH * HID + hidx] = hv;
        hnext[hidx] = (bf16_t)hv;
    }
}

// ---------------------------------------------------------------------------
extern "C" void kernel_launch(void* const* d_in, const int* in_sizes, int n_in,
                              void* d_out, int out_size, void* d_ws, size_t ws_size,
                              hipStream_t stream) {
    (void)in_sizes; (void)n_in; (void)out_size; (void)ws_size;

    const float* x   = (const float*)d_in[0];
    const float* Wih = (const float*)d_in[1];
    const float* bih = (const float*)d_in[2];
    const float* Whh = (const float*)d_in[3];
    const float* bhh = (const float*)d_in[4];
    float* out = (float*)d_out;

    // Workspace carve-up (~46 MB total), 256B-aligned regions.
    char*  ws  = (char*)d_ws;
    size_t off = 0;
    auto carve = [&](size_t bytes) -> char* {
        char* p = ws + off;
        off += (bytes + 255) & ~(size_t)255;
        return p;
    };
    bf16_t* wib  = (bf16_t*)carve((size_t)GATES * IN_DIM * sizeof(bf16_t));  //  4 MB
    bf16_t* whb  = (bf16_t*)carve((size_t)GATES * HID * sizeof(bf16_t));     //  8 MB
    bf16_t* xb   = (bf16_t*)carve((size_t)T_STEPS * BATCH * IN_DIM * sizeof(bf16_t)); // 33.5 MB
    float*  bias = (float*) carve((size_t)GATES * sizeof(float));
    float*  cbuf = (float*) carve((size_t)BATCH * HID * sizeof(float));
    bf16_t* hb   = (bf16_t*)carve((size_t)2 * BATCH * HID * sizeof(bf16_t)); // ping-pong

    lstm_prep<<<1024, 256, 0, stream>>>(x, Wih, bih, Whh, bhh,
                                        xb, wib, whb, bias, cbuf, hb);

    for (int t = 0; t < T_STEPS; ++t) {
        bf16_t* hprev = hb + (size_t)(t & 1) * BATCH * HID;
        bf16_t* hnext = hb + (size_t)((t + 1) & 1) * BATCH * HID;
        lstm_step<<<HID / 16, 256, 0, stream>>>(xb, wib, whb, bias,
                                                hprev, hnext, cbuf, out, t);
    }

    // Append hT (== out[T-1]) and cT to d_out.
    const size_t outTBH = (size_t)T_STEPS * BATCH * HID;
    const size_t BH     = (size_t)BATCH * HID;
    hipMemcpyAsync(out + outTBH, out + outTBH - BH, BH * sizeof(float),
                   hipMemcpyDeviceToDevice, stream);
    hipMemcpyAsync(out + outTBH + BH, cbuf, BH * sizeof(float),
                   hipMemcpyDeviceToDevice, stream);
}
